// MultiHeadAttentionV1_89137751261756
// MI455X (gfx1250) — matrix operand order repaired and checked by reading, not compiled, not run on hardware
//
#include <hip/hip_runtime.h>
#include <hip/hip_bf16.h>

// MI455X / gfx1250 multi-head attention, f16 WMMA pipeline with f32 accumulate.
// Shapes: B=2, S=2048, E=1024, H=16, D=64.

typedef __attribute__((ext_vector_type(16))) _Float16 v16h;
typedef __attribute__((ext_vector_type(8)))  float    v8f;
typedef __attribute__((ext_vector_type(4)))  int      v4i;

#define NB   2
#define NS   2048
#define NE   1024
#define NH   16
#define ND   64

#if defined(__has_builtin)
#  if __has_builtin(__builtin_amdgcn_global_load_async_to_lds_b128)
#    define HAS_ASYNC_LDS 1
#  else
#    define HAS_ASYNC_LDS 0
#  endif
#  if __has_builtin(__builtin_amdgcn_s_wait_asynccnt)
#    define HAS_WAIT_ASYNC 1
#  else
#    define HAS_WAIT_ASYNC 0
#  endif
#else
#  define HAS_ASYNC_LDS 0
#  define HAS_WAIT_ASYNC 0
#endif

// 16-byte global -> LDS copy; async (ASYNCcnt) when the toolchain exposes it.
static __device__ inline void copy_b128_to_lds(const _Float16* g, _Float16* l) {
#if HAS_ASYNC_LDS
  __builtin_amdgcn_global_load_async_to_lds_b128((v4i*)g, (v4i*)l, 0, 0);
#else
  *(uint4*)l = *(const uint4*)g;
#endif
}

static __device__ inline void wait_async_le2() {
#if HAS_ASYNC_LDS
#  if HAS_WAIT_ASYNC
  __builtin_amdgcn_s_wait_asynccnt(2);
#  else
  asm volatile("s_wait_asynccnt 0x2" ::: "memory");
#  endif
#endif
}

static __device__ inline void wait_async_le0() {
#if HAS_ASYNC_LDS
#  if HAS_WAIT_ASYNC
  __builtin_amdgcn_s_wait_asynccnt(0);
#  else
  asm volatile("s_wait_asynccnt 0x0" ::: "memory");
#  endif
#endif
}

static __device__ inline v8f wmma_f16(v16h a, v16h b, v8f c) {
  return __builtin_amdgcn_wmma_f32_16x16x32_f16(
      /*neg_a=*/false, a, /*neg_b=*/false, b,
      /*c_mod=*/(short)0, c, /*reuse_a=*/false, /*reuse_b=*/false);
}

// A-fragment (16x32 f16), lane m=lane_lo holds row m.
// hi==0: halves[0..7]=K0..7, halves[8..15]=K16..23
// hi==1: halves[0..7]=K8..15, halves[8..15]=K24..31
static __device__ inline v16h a_frag_f16(const _Float16* __restrict__ p, int hi) {
  v16h a;
#pragma unroll
  for (int j = 0; j < 8; ++j) {
    a[j]     = p[hi * 8 + j];
    a[8 + j] = p[16 + hi * 8 + j];
  }
  return a;
}

static __device__ inline v16h a_frag_f32(const float* __restrict__ p, int hi) {
  v16h a;
#pragma unroll
  for (int j = 0; j < 8; ++j) {
    a[j]     = (_Float16)p[hi * 8 + j];
    a[8 + j] = (_Float16)p[16 + hi * 8 + j];
  }
  return a;
}

// B-fragment (32x16 f16), lane n=lane_lo holds column n; p points at Bt[n][k0]
// (column-major staging), halves[j] = K = hi*8 + j, halves[8+j] = K = 16 + hi*8 + j.
static __device__ inline v16h b_frag_f16(const _Float16* __restrict__ p, int hi) {
  v16h b;
#pragma unroll
  for (int j = 0; j < 8; ++j) { b[j] = p[hi * 8 + j]; b[8 + j] = p[16 + hi * 8 + j]; }
  return b;
}

// ---------------------------------------------------------------------------
// Kernel 1: per-head projection  Out[bh, s, d] = X[b, s, :] @ W[h, :, d] + bias
// grid: (S/128, B*H), block 256 (8 waves x 16 rows)
// ---------------------------------------------------------------------------
__global__ __launch_bounds__(256) void qkv_proj_kernel(
    const float* __restrict__ X,      // [B, S, E]
    const float* __restrict__ W,      // [H, E, D]
    const float* __restrict__ bias,   // [H, D]
    _Float16* __restrict__ Out) {     // [B*H, S, D] f16
  __shared__ alignas(16) _Float16 sWt[ND * 32];  // Wt[d][k], transposed tile

  const int tid     = threadIdx.x;
  const int wave    = tid >> 5;
  const int lane    = tid & 31;
  const int lane_lo = lane & 15;
  const int hi      = lane >> 4;

  const int bh   = blockIdx.y;
  const int b    = bh >> 4;
  const int h    = bh & 15;
  const int row0 = blockIdx.x * 128;

  const float* xrow = X + ((size_t)b * NS + (row0 + wave * 16 + lane_lo)) * NE;
  const float* Wh   = W + (size_t)h * NE * ND;

  v8f zero = {};
  v8f acc[4];
#pragma unroll
  for (int nt = 0; nt < 4; ++nt) acc[nt] = zero;

  for (int kc = 0; kc < NE; kc += 32) {
    __syncthreads();
    for (int i = tid; i < ND * 32; i += 256) {
      int k = i >> 6, d = i & 63;
      sWt[d * 32 + k] = (_Float16)Wh[(size_t)(kc + k) * ND + d];
    }
    __syncthreads();

    v16h af = a_frag_f32(xrow + kc, hi);
#pragma unroll
    for (int nt = 0; nt < 4; ++nt) {
      v16h bf = b_frag_f16(&sWt[(nt * 16 + lane_lo) * 32], hi);
      acc[nt] = wmma_f16(af, bf, acc[nt]);
    }
  }

#pragma unroll
  for (int nt = 0; nt < 4; ++nt) {
    float bv = bias[h * ND + nt * 16 + lane_lo];
#pragma unroll
    for (int r = 0; r < 8; ++r) {
      int row = row0 + wave * 16 + r + 8 * hi;
      Out[((size_t)bh * NS + row) * ND + nt * 16 + lane_lo] =
          (_Float16)(acc[nt][r] + bv);
    }
  }
}

// ---------------------------------------------------------------------------
// Kernel 2: flash attention per (b,h).  grid: (S/128, B*H), block 256 (8 waves)
// K tiles double-buffered via async global->LDS (ASYNCcnt); V transposed via
// VGPRs.  Writes ctx f16 directly in concat layout [B, S, H*D].
// ---------------------------------------------------------------------------
__global__ __launch_bounds__(256) void attn_kernel(
    const _Float16* __restrict__ Q,   // [B*H, S, D]
    const _Float16* __restrict__ K,   // [B*H, S, D]
    const _Float16* __restrict__ V,   // [B*H, S, D]
    _Float16* __restrict__ Ctx) {     // [B, S, H*D]
  __shared__ alignas(16) _Float16 sK[2 * 64 * ND];  // double-buffered [krow][d]
  __shared__ alignas(16) _Float16 sVt[ND * 64];     // [d][krow] transposed
  __shared__ alignas(16) _Float16 sP[8 * 16 * 64];  // per-wave P tile [16][64]

  const int tid     = threadIdx.x;
  const int wave    = tid >> 5;
  const int lane    = tid & 31;
  const int lane_lo = lane & 15;
  const int hi      = lane >> 4;

  const int bh = blockIdx.y;
  const int b  = bh >> 4;
  const int h  = bh & 15;
  const int q0 = blockIdx.x * 128;
  const float scale = 0.125f;  // 1/sqrt(64)

  // Q fragments for this wave's 16 rows (K-dim = D = 64 -> two 32-chunks)
  const _Float16* qrow = Q + ((size_t)bh * NS + (q0 + wave * 16 + lane_lo)) * ND;
  v16h qf[2];
#pragma unroll
  for (int kc2 = 0; kc2 < 2; ++kc2) qf[kc2] = a_frag_f16(qrow + kc2 * 32, hi);

  v8f zero = {};
  v8f acc[4];
#pragma unroll
  for (int nt = 0; nt < 4; ++nt) acc[nt] = zero;
  float mrow[8], lrow[8];
#pragma unroll
  for (int r = 0; r < 8; ++r) { mrow[r] = -3.0e38f; lrow[r] = 0.0f; }

  _Float16* Pw = sP + wave * 16 * 64;
  const int NT = NS / 64;  // k tiles

  // Prologue: kick off async copy of K tile 0 (2 async instr per wave).
  {
    const _Float16* ksrc = K + ((size_t)bh * NS) * ND;
    for (int i = tid; i < 512; i += 256)
      copy_b128_to_lds(ksrc + i * 8, sK + i * 8);
  }

  for (int kt = 0; kt < NT; ++kt) {
    const int cur = kt & 1;
    const _Float16* vsrc = V + ((size_t)bh * NS + kt * 64) * ND;

    __syncthreads();  // prior compute done: sVt + the other K buffer are free

    // V tile: transpose into sVt[d][krow] through VGPRs
    for (int i = tid; i < 512; i += 256) {
      int kr = i >> 3;
      int dc = (i & 7) * 8;
      union { uint4 u; _Float16 hh[8]; } t;
      t.u = *(const uint4*)(vsrc + (size_t)kr * ND + dc);
#pragma unroll
      for (int j = 0; j < 8; ++j) sVt[(dc + j) * 64 + kr] = t.hh[j];
    }

    // Kick off async copy of the NEXT K tile into the other buffer, then
    // retire the current tile (async loads complete in order; <=2 outstanding
    // means the 2 copies of the current tile have landed).
    if (kt + 1 < NT) {
      const _Float16* knext = K + ((size_t)bh * NS + (kt + 1) * 64) * ND;
      _Float16* dst = sK + (cur ^ 1) * (64 * ND);
      for (int i = tid; i < 512; i += 256)
        copy_b128_to_lds(knext + i * 8, dst + i * 8);
      __builtin_prefetch(V + ((size_t)bh * NS + (kt + 1) * 64) * ND, 0, 1);
      wait_async_le2();
    } else {
      wait_async_le0();
    }
    __syncthreads();

    const _Float16* sKc = sK + cur * (64 * ND);

    // scores S = Q @ K^T  (B-frag: lane = local k-row, contiguous d in sKc)
    v8f cs[4];
#pragma unroll
    for (int nt = 0; nt < 4; ++nt) {
      v8f c = zero;
#pragma unroll
      for (int kc2 = 0; kc2 < 2; ++kc2) {
        v16h bf = b_frag_f16(&sKc[(nt * 16 + lane_lo) * 64 + kc2 * 32], hi);
        c = wmma_f16(qf[kc2], bf, c);
      }
      cs[nt] = c;
    }

    // online softmax; slot r on this lane is row (r + 8*hi)
    float tmax[8];
#pragma unroll
    for (int r = 0; r < 8; ++r) {
      float m = -3.0e38f;
#pragma unroll
      for (int nt = 0; nt < 4; ++nt) {
        float v = cs[nt][r] * scale;
        cs[nt][r] = v;
        m = fmaxf(m, v);
      }
      tmax[r] = m;
    }
#pragma unroll
    for (int r = 0; r < 8; ++r) {
#pragma unroll
      for (int mask = 1; mask < 16; mask <<= 1)
        tmax[r] = fmaxf(tmax[r], __shfl_xor(tmax[r], mask, 32));
    }

    float rs[8];
#pragma unroll
    for (int r = 0; r < 8; ++r) {
      float nm = fmaxf(mrow[r], tmax[r]);
      float al = __expf(mrow[r] - nm);
      mrow[r] = nm;
      float s = 0.0f;
#pragma unroll
      for (int nt = 0; nt < 4; ++nt) {
        float p = __expf(cs[nt][r] - nm);
        cs[nt][r] = p;
        s += p;
      }
      rs[r] = s;
#pragma unroll
      for (int nt = 0; nt < 4; ++nt) acc[nt][r] *= al;
      lrow[r] *= al;
    }
#pragma unroll
    for (int r = 0; r < 8; ++r) {
#pragma unroll
      for (int mask = 1; mask < 16; mask <<= 1)
        rs[r] += __shfl_xor(rs[r], mask, 32);
      lrow[r] += rs[r];
    }

    // P -> per-wave LDS (C layout -> row-major), then ctx += P @ V
#pragma unroll
    for (int nt = 0; nt < 4; ++nt)
#pragma unroll
      for (int r = 0; r < 8; ++r)
        Pw[(r + 8 * hi) * 64 + nt * 16 + lane_lo] = (_Float16)cs[nt][r];

    v16h pf[2];
#pragma unroll
    for (int kc2 = 0; kc2 < 2; ++kc2)
      pf[kc2] = a_frag_f16(Pw + lane_lo * 64 + kc2 * 32, hi);

#pragma unroll
    for (int nt = 0; nt < 4; ++nt) {
#pragma unroll
      for (int kc2 = 0; kc2 < 2; ++kc2) {
        v16h vf = b_frag_f16(&sVt[(nt * 16 + lane_lo) * 64 + kc2 * 32], hi);
        acc[nt] = wmma_f16(pf[kc2], vf, acc[nt]);
      }
    }
  }

  // normalize and store ctx in concat layout
#pragma unroll
  for (int nt = 0; nt < 4; ++nt) {
#pragma unroll
    for (int r = 0; r < 8; ++r) {
      int   qr = q0 + wave * 16 + r + 8 * hi;
      float v  = acc[nt][r] / lrow[r];
      Ctx[((size_t)b * NS + qr) * NE + h * ND + nt * 16 + lane_lo] = (_Float16)v;
    }
  }
}

// ---------------------------------------------------------------------------
// Kernel 3: output projection  Out = Ctx[BS, E] @ Wo[E, E] + bo
// grid: (B*S/128, E/64), block 256 (8 waves)
// ---------------------------------------------------------------------------
__global__ __launch_bounds__(256) void out_proj_kernel(
    const _Float16* __restrict__ Ctx,  // [B*S, E] f16
    const float* __restrict__ Wo,      // [E, E]
    const float* __restrict__ bo,      // [E]
    float* __restrict__ Out) {         // [B*S, E]
  __shared__ alignas(16) _Float16 sWt[64 * 32];  // Wt[n][k]

  const int tid     = threadIdx.x;
  const int wave    = tid >> 5;
  const int lane    = tid & 31;
  const int lane_lo = lane & 15;
  const int hi      = lane >> 4;

  const int row0 = blockIdx.x * 128;
  const int n0   = blockIdx.y * 64;

  const _Float16* arow = Ctx + (size_t)(row0 + wave * 16 + lane_lo) * NE;

  v8f zero = {};
  v8f acc[4];
#pragma unroll
  for (int nt = 0; nt < 4; ++nt) acc[nt] = zero;

  for (int kc = 0; kc < NE; kc += 32) {
    __syncthreads();
    for (int i = tid; i < 64 * 32; i += 256) {
      int k = i >> 6, n = i & 63;
      sWt[n * 32 + k] = (_Float16)Wo[(size_t)(kc + k) * NE + n0 + n];
    }
    __syncthreads();

    v16h af = a_frag_f16(arow + kc, hi);
#pragma unroll
    for (int nt = 0; nt < 4; ++nt) {
      v16h bf = b_frag_f16(&sWt[(nt * 16 + lane_lo) * 32], hi);
      acc[nt] = wmma_f16(af, bf, acc[nt]);
    }
  }

#pragma unroll
  for (int nt = 0; nt < 4; ++nt) {
    float bv = bo[n0 + nt * 16 + lane_lo];
#pragma unroll
    for (int r = 0; r < 8; ++r) {
      int row = row0 + wave * 16 + r + 8 * hi;
      Out[(size_t)row * NE + n0 + nt * 16 + lane_lo] = acc[nt][r] + bv;
    }
  }
}

// ---------------------------------------------------------------------------
extern "C" void kernel_launch(void* const* d_in, const int* in_sizes, int n_in,
                              void* d_out, int out_size, void* d_ws, size_t ws_size,
                              hipStream_t stream) {
  const float* q  = (const float*)d_in[0];
  const float* k  = (const float*)d_in[1];
  const float* v  = (const float*)d_in[2];
  const float* Wq = (const float*)d_in[3];
  const float* bq = (const float*)d_in[4];
  const float* Wk = (const float*)d_in[5];
  const float* bk = (const float*)d_in[6];
  const float* Wv = (const float*)d_in[7];
  const float* bv = (const float*)d_in[8];
  const float* Wo = (const float*)d_in[9];
  const float* bo = (const float*)d_in[10];

  // Workspace layout (f16): Q, K, V = [B*H, S, D] each; Ctx = [B, S, E]
  const size_t qkv_bytes = (size_t)NB * NH * NS * ND * sizeof(_Float16);  // 8 MB
  char* ws = (char*)d_ws;
  _Float16* Qb  = (_Float16*)(ws);
  _Float16* Kb  = (_Float16*)(ws + qkv_bytes);
  _Float16* Vb  = (_Float16*)(ws + 2 * qkv_bytes);
  _Float16* Ctx = (_Float16*)(ws + 3 * qkv_bytes);

  dim3 gProj(NS / 128, NB * NH);
  qkv_proj_kernel<<<gProj, 256, 0, stream>>>(q, Wq, bq, Qb);
  qkv_proj_kernel<<<gProj, 256, 0, stream>>>(k, Wk, bk, Kb);
  qkv_proj_kernel<<<gProj, 256, 0, stream>>>(v, Wv, bv, Vb);

  dim3 gAttn(NS / 128, NB * NH);
  attn_kernel<<<gAttn, 256, 0, stream>>>(Qb, Kb, Vb, Ctx);

  dim3 gOut(NB * NS / 128, NE / 64);
  out_proj_kernel<<<gOut, 256, 0, stream>>>(Ctx, Wo, bo, (float*)d_out);
}
